// S6_fusion_24876450578939
// MI455X (gfx1250) — matrix operand order repaired and checked by reading, not compiled, not run on hardware
//
#include <hip/hip_runtime.h>

// ---------------------------------------------------------------------------
// S6 (Mamba selective scan) for MI455X / gfx1250, wave32.
//  Kernel 1: fused projections, v_wmma_f32_16x16x32_bf16 (f32 accum),
//            software-pipelined global->LDS; A and B fragments both read as
//            ds_load_b128 thanks to transposed, padded LDS tiles.
//  Kernels 2-4: chunked linear-recurrence scan (decay = exp(A_n * sum(delta))).
// ---------------------------------------------------------------------------

#define B_ 4
#define D_ 1024
#define L_ 2048
#define N_ 16
#define R_ 64
#define NCH 16            // scan chunks
#define CHL (L_ / NCH)    // 128 timesteps per chunk

#define APAD 40           // activation LDS row stride (bf16): 80B, 16B-aligned
#define DTPAD 72          // dt_low LDS row stride (bf16): 144B, 16B-aligned

typedef __attribute__((ext_vector_type(16))) __bf16 v16bf;
typedef __attribute__((ext_vector_type(2)))  __bf16 v2bf;
typedef __attribute__((ext_vector_type(8)))  float  v8f;

__device__ __forceinline__ float softplus_f(float v) {
    return v > 20.f ? v : log1pf(__expf(v));
}

__device__ __forceinline__ unsigned pack_bf16(float a, float b) {
    union { unsigned u; __bf16 h[2]; } c;
    c.h[0] = (__bf16)a;
    c.h[1] = (__bf16)b;
    return c.u;   // v_cvt_pk_bf16_f32
}

// ---------------------------------------------------------------------------
// Kernel 1: projections.  Grid: (L/16, B), 256 threads (8 waves).
// ---------------------------------------------------------------------------
__global__ __launch_bounds__(256)
void s6_proj_kernel(const float* __restrict__ u,
                    const float* __restrict__ mask,
                    const float* __restrict__ info,
                    const float* __restrict__ W_C,
                    const float* __restrict__ W_B,
                    const float* __restrict__ W_delta,
                    const float* __restrict__ W_dt,
                    const float* __restrict__ dt_bias,
                    float* __restrict__ dd,       // (B,L,D,2): delta, delta*u
                    float* __restrict__ BmT,      // (B,L,N)
                    float* __restrict__ CmT)      // (B,L,N)
{
    const int tid  = threadIdx.x;
    const int lane = tid & 31;
    const int wave = tid >> 5;
    const int hf   = lane >> 4;    // half-wave (K-split in WMMA layouts)
    const int m    = lane & 15;
    const int l0   = blockIdx.x * 16;
    const int b    = blockIdx.y;

    __shared__ __bf16 sWd[R_][32];          // W_delta A-slab [M=r][K=d]
    __shared__ __bf16 sWB[N_][32];
    __shared__ __bf16 sWC[N_][32];
    __shared__ __bf16 sMaskT[16][APAD];     // activations TRANSPOSED [N=l][K=d]
    __shared__ __bf16 sInfoT[16][APAD];
    __shared__ __bf16 sUT[16][APAD];
    __shared__ __bf16 sDtLowT[16][DTPAD];   // dt_low TRANSPOSED [N=l][K=r]

    unsigned* const uWd   = (unsigned*)sWd;
    unsigned* const uWB   = (unsigned*)sWB;
    unsigned* const uWC   = (unsigned*)sWC;
    unsigned* const uMask = (unsigned*)sMaskT;
    unsigned* const uInfo = (unsigned*)sInfoT;
    unsigned* const uUu   = (unsigned*)sUT;

    // gather staging
    float2 gWd[4], gWB, gWC;
    float  gM0, gM1, gI0, gI1, gU0, gU1;
    const int lp = tid & 15;       // l within tile (coalesced global runs)
    const int dp = tid >> 4;       // d-pair index 0..15

    auto gather = [&](int kc) {
        #pragma unroll
        for (int q = 0; q < 4; ++q) {              // W_delta 64x32 = 1024 pairs
            int e = tid + 256 * q;
            int r = e >> 4, kp = e & 15;
            gWd[q] = *(const float2*)&W_delta[r * D_ + kc + kp * 2];
        }
        {   int r = tid >> 4, kp = tid & 15;
            gWB = *(const float2*)&W_B[r * D_ + kc + kp * 2];
            gWC = *(const float2*)&W_C[r * D_ + kc + kp * 2];
        }
        {   // activation d-pair at one l (transposed commit target)
            long base = ((long)b * D_ + kc + 2 * dp) * L_ + l0 + lp;
            gM0 = mask[base]; gM1 = mask[base + L_];
            gI0 = info[base]; gI1 = info[base + L_];
            gU0 = u[base];    gU1 = u[base + L_];
        }
    };
    auto commit = [&]() {
        #pragma unroll
        for (int q = 0; q < 4; ++q) {
            int e = tid + 256 * q;
            int r = e >> 4, kp = e & 15;
            uWd[r * 16 + kp] = pack_bf16(gWd[q].x, gWd[q].y);
        }
        {   int r = tid >> 4, kp = tid & 15;
            uWB[r * 16 + kp] = pack_bf16(gWB.x, gWB.y);
            uWC[r * 16 + kp] = pack_bf16(gWC.x, gWC.y);
        }
        {   int ui = lp * (APAD / 2) + dp;      // row stride 20 uints
            uMask[ui] = pack_bf16(gM0, gM1);
            uInfo[ui] = pack_bf16(gI0, gI1);
            uUu[ui]   = pack_bf16(gU0, gU1);
        }
    };

    v8f acc = {};  // waves 0-3: dt_low tiles; wave 4: Bm; wave 5: Cm
    gather(0);

    for (int kc = 0; kc < D_; kc += 32) {
        __syncthreads();
        commit();
        if (kc + 32 < D_) gather(kc + 32);   // next-slab loads fly over WMMAs
        __syncthreads();

        if (wave < 6) {
            const __bf16* arow;
            const __bf16* brow;
            if (wave < 4)       { arow = &sWd[wave * 16 + m][0]; brow = &sMaskT[m][0]; }
            else if (wave == 4) { arow = &sWB[m][0];             brow = &sInfoT[m][0]; }
            else                { arow = &sWC[m][0];             brow = &sUT[m][0];    }

            v16bf av, bv;
            // A fragment in contiguous k order (as B): half 0 holds K 0..15, half 1 holds K 16..31.
            #pragma unroll
            for (int t = 0; t < 8; ++t) {
                int i = 2 * t;
                int k = (hf << 4) + i;   // even
                v2bf w = *(const v2bf*)&arow[k];
                av[i]     = w.x;
                av[i + 1] = w.y;
            }
            // B fragment: 32 contiguous bytes -> 2x ds_load_b128
            const v2bf* bp = (const v2bf*)&brow[hf * 16];
            #pragma unroll
            for (int t = 0; t < 8; ++t) {
                bv[2 * t]     = bp[t].x;
                bv[2 * t + 1] = bp[t].y;
            }
            acc = __builtin_amdgcn_wmma_f32_16x16x32_bf16(
                      false, av, false, bv, (short)0, acc, false, false);
        }
    }

    __syncthreads();
    // C/D layout: vgpr j holds M = j + 8*hf, N = m.
    if (wave < 4) {
        #pragma unroll
        for (int j = 0; j < 8; ++j)
            sDtLowT[m][wave * 16 + j + 8 * hf] = (__bf16)acc[j];
    } else if (wave == 4) {
        #pragma unroll
        for (int j = 0; j < 8; ++j)
            BmT[((long)b * L_ + l0 + m) * N_ + j + 8 * hf] = acc[j];
    } else if (wave == 5) {
        #pragma unroll
        for (int j = 0; j < 8; ++j)
            CmT[((long)b * L_ + l0 + m) * N_ + j + 8 * hf] = acc[j];
    }
    __syncthreads();

    // ---- Phase 2: delta = softplus(W_dt(1024x64) @ dt_low(64x16) + bias) ----
    for (int t = wave; t < D_ / 16; t += 8) {
        int d0t = t * 16;
        const float* wrow = &W_dt[(d0t + m) * R_];
        v8f c2 = {};
        #pragma unroll
        for (int kc2 = 0; kc2 < R_; kc2 += 32) {
            float4 f0 = *(const float4*)&wrow[kc2 + hf * 16];
            float4 f1 = *(const float4*)&wrow[kc2 + hf * 16 + 4];
            float4 f2 = *(const float4*)&wrow[kc2 + hf * 16 + 8];
            float4 f3 = *(const float4*)&wrow[kc2 + hf * 16 + 12];
            v16bf av, bv;
            av[0]=(__bf16)f0.x;  av[1]=(__bf16)f0.y;  av[2]=(__bf16)f0.z;  av[3]=(__bf16)f0.w;
            av[4]=(__bf16)f1.x;  av[5]=(__bf16)f1.y;  av[6]=(__bf16)f1.z;  av[7]=(__bf16)f1.w;
            av[8]=(__bf16)f2.x;  av[9]=(__bf16)f2.y;  av[10]=(__bf16)f2.z; av[11]=(__bf16)f2.w;
            av[12]=(__bf16)f3.x; av[13]=(__bf16)f3.y; av[14]=(__bf16)f3.z; av[15]=(__bf16)f3.w;
            const v2bf* bp = (const v2bf*)&sDtLowT[m][kc2 + hf * 16];
            #pragma unroll
            for (int q = 0; q < 8; ++q) {
                bv[2 * q]     = bp[q].x;
                bv[2 * q + 1] = bp[q].y;
            }
            c2 = __builtin_amdgcn_wmma_f32_16x16x32_bf16(
                     false, av, false, bv, (short)0, c2, false, false);
        }
        int l = l0 + m;
        float4 bb0 = *(const float4*)&dt_bias[d0t + 8 * hf];
        float4 bb1 = *(const float4*)&dt_bias[d0t + 8 * hf + 4];
        float bias8[8] = {bb0.x, bb0.y, bb0.z, bb0.w, bb1.x, bb1.y, bb1.z, bb1.w};
        float uv8[8];
        #pragma unroll
        for (int j = 0; j < 8; ++j)
            uv8[j] = u[((long)b * D_ + d0t + j + 8 * hf) * L_ + l];
        float dl8[8];
        #pragma unroll
        for (int j = 0; j < 8; ++j)
            dl8[j] = softplus_f(c2[j] + bias8[j]);
        long base2 = (((long)b * L_ + l) * D_ + d0t + 8 * hf) * 2;
        #pragma unroll
        for (int jj = 0; jj < 4; ++jj) {
            float4 st = {dl8[2*jj],     dl8[2*jj]     * uv8[2*jj],
                         dl8[2*jj + 1], dl8[2*jj + 1] * uv8[2*jj + 1]};
            *(float4*)&dd[base2 + 4 * jj] = st;
        }
    }
}

// ---------------------------------------------------------------------------
// Kernel 2: per-chunk local scan (x from 0) + sum(delta).
// Grid: (D/256, B, NCH), 256 threads; lane = one d.
// ---------------------------------------------------------------------------
__global__ __launch_bounds__(256)
void s6_scan_partial(const float* __restrict__ dd,     // (B,L,D,2)
                     const float* __restrict__ BmT,
                     const float* __restrict__ A_log,
                     float* __restrict__ chunkState,   // (B,NCH,D,N)
                     float* __restrict__ chunkSum)     // (B,NCH,D)
{
    const int tid = threadIdx.x;
    const int d   = blockIdx.x * 256 + tid;
    const int b   = blockIdx.y;
    const int c   = blockIdx.z;
    const int lbase = c * CHL;
    const float2* dd2 = (const float2*)dd;

    float An[N_];
    #pragma unroll
    for (int q = 0; q < 4; ++q) {
        float4 t = *(const float4*)&A_log[d * N_ + q * 4];
        An[q*4+0] = -__expf(t.x); An[q*4+1] = -__expf(t.y);
        An[q*4+2] = -__expf(t.z); An[q*4+3] = -__expf(t.w);
    }
    float x[N_];
    #pragma unroll
    for (int n = 0; n < N_; ++n) x[n] = 0.f;
    float sumD = 0.f;

    __shared__ float sB[32][N_];

    for (int ls = 0; ls < CHL; ls += 32) {
        __syncthreads();
        if (tid < 128) {
            int lc = tid >> 2, nq = (tid & 3) * 4;
            *(float4*)&sB[lc][nq] =
                *(const float4*)&BmT[((long)b * L_ + lbase + ls + lc) * N_ + nq];
        }
        __syncthreads();
        const float2* p = &dd2[((long)b * L_ + lbase + ls) * D_ + d];
        for (int lc = 0; lc < 32; ++lc) {
            // prefetch ~8 timesteps ahead (each step is a fresh 2KB slab/block)
            long lpf = lbase + ls + lc + 8;
            __builtin_prefetch(&dd2[(((long)b * L_) + (lpf < L_ ? lpf : L_ - 1)) * D_ + d], 0, 1);
            float2 vd = *p; p += D_;
            sumD += vd.x;
            #pragma unroll
            for (int n = 0; n < N_; ++n)
                x[n] = __expf(vd.x * An[n]) * x[n] + vd.y * sB[lc][n];
        }
    }
    long so = (long)(b * NCH + c) * D_ + d;
    #pragma unroll
    for (int q = 0; q < 4; ++q) {
        float4 st = {x[q*4+0], x[q*4+1], x[q*4+2], x[q*4+3]};
        *(float4*)&chunkState[so * N_ + q * 4] = st;
    }
    chunkSum[so] = sumD;
}

// ---------------------------------------------------------------------------
// Kernel 3: combine chunk carries. Chunk decay P_n = exp(A_n * sum(delta)).
// Grid: (D/256, B).
// ---------------------------------------------------------------------------
__global__ __launch_bounds__(256)
void s6_scan_combine(const float* __restrict__ chunkState,
                     const float* __restrict__ chunkSum,
                     const float* __restrict__ A_log,
                     float* __restrict__ initState)    // (B,NCH,D,N)
{
    const int tid = threadIdx.x;
    const int d   = blockIdx.x * 256 + tid;
    const int b   = blockIdx.y;

    float An[N_];
    #pragma unroll
    for (int q = 0; q < 4; ++q) {
        float4 t = *(const float4*)&A_log[d * N_ + q * 4];
        An[q*4+0] = -__expf(t.x); An[q*4+1] = -__expf(t.y);
        An[q*4+2] = -__expf(t.z); An[q*4+3] = -__expf(t.w);
    }
    float carry[N_];
    #pragma unroll
    for (int n = 0; n < N_; ++n) carry[n] = 0.f;

    for (int c = 0; c < NCH; ++c) {
        long so = (long)(b * NCH + c) * D_ + d;
        #pragma unroll
        for (int q = 0; q < 4; ++q) {
            float4 st = {carry[q*4+0], carry[q*4+1], carry[q*4+2], carry[q*4+3]};
            *(float4*)&initState[so * N_ + q * 4] = st;
        }
        float sd = chunkSum[so];
        #pragma unroll
        for (int q = 0; q < 4; ++q) {
            float4 cs = *(const float4*)&chunkState[so * N_ + q * 4];
            carry[q*4+0] = __expf(An[q*4+0] * sd) * carry[q*4+0] + cs.x;
            carry[q*4+1] = __expf(An[q*4+1] * sd) * carry[q*4+1] + cs.y;
            carry[q*4+2] = __expf(An[q*4+2] * sd) * carry[q*4+2] + cs.z;
            carry[q*4+3] = __expf(An[q*4+3] * sd) * carry[q*4+3] + cs.w;
        }
    }
}

// ---------------------------------------------------------------------------
// Kernel 4: replay chunk with true initial state, emit y via padded LDS,
// float4-coalesced stores along l (output layout (B,D,L)).
// Grid: (D/256, B, NCH).
// ---------------------------------------------------------------------------
__global__ __launch_bounds__(256)
void s6_scan_final(const float* __restrict__ dd,
                   const float* __restrict__ BmT,
                   const float* __restrict__ CmT,
                   const float* __restrict__ A_log,
                   const float* __restrict__ initState,
                   float* __restrict__ out)           // (B,D,L)
{
    const int tid = threadIdx.x;
    const int d0  = blockIdx.x * 256;
    const int d   = d0 + tid;
    const int b   = blockIdx.y;
    const int c   = blockIdx.z;
    const int lbase = c * CHL;
    const float2* dd2 = (const float2*)dd;

    float An[N_], x[N_];
    #pragma unroll
    for (int q = 0; q < 4; ++q) {
        float4 t = *(const float4*)&A_log[d * N_ + q * 4];
        An[q*4+0] = -__expf(t.x); An[q*4+1] = -__expf(t.y);
        An[q*4+2] = -__expf(t.z); An[q*4+3] = -__expf(t.w);
    }
    long so = (long)(b * NCH + c) * D_ + d;
    #pragma unroll
    for (int q = 0; q < 4; ++q) {
        float4 t = *(const float4*)&initState[so * N_ + q * 4];
        x[q*4+0] = t.x; x[q*4+1] = t.y; x[q*4+2] = t.z; x[q*4+3] = t.w;
    }

    __shared__ float sB[32][N_];
    __shared__ float sC[32][N_];
    __shared__ float ytile[256 * 33];   // stride 33: conflict-free b32 traffic

    for (int ls = 0; ls < CHL; ls += 32) {
        __syncthreads();
        {   int e = tid & 127;
            int lc = e >> 2, nq = (e & 3) * 4;
            long bo = ((long)b * L_ + lbase + ls + lc) * N_ + nq;
            if (tid < 128) *(float4*)&sB[lc][nq] = *(const float4*)&BmT[bo];
            else           *(float4*)&sC[lc][nq] = *(const float4*)&CmT[bo];
        }
        __syncthreads();
        const float2* p = &dd2[((long)b * L_ + lbase + ls) * D_ + d];
        for (int lc = 0; lc < 32; ++lc) {
            long lpf = lbase + ls + lc + 8;
            __builtin_prefetch(&dd2[(((long)b * L_) + (lpf < L_ ? lpf : L_ - 1)) * D_ + d], 0, 1);
            float2 vd = *p; p += D_;
            float y = 0.f;
            #pragma unroll
            for (int n = 0; n < N_; ++n) {
                x[n] = __expf(vd.x * An[n]) * x[n] + vd.y * sB[lc][n];
                y += x[n] * sC[lc][n];
            }
            ytile[tid * 33 + lc] = y;
        }
        __syncthreads();
        #pragma unroll
        for (int g = 0; g < 8; ++g) {    // 8 lanes x float4 = one 32-l row
            int dl  = g * 32 + (tid >> 3);
            int lc4 = (tid & 7) * 4;
            float4 y4 = {ytile[dl * 33 + lc4],     ytile[dl * 33 + lc4 + 1],
                         ytile[dl * 33 + lc4 + 2], ytile[dl * 33 + lc4 + 3]};
            *(float4*)&out[((long)b * D_ + d0 + dl) * L_ + lbase + ls + lc4] = y4;
        }
    }
}

// ---------------------------------------------------------------------------
extern "C" void kernel_launch(void* const* d_in, const int* in_sizes, int n_in,
                              void* d_out, int out_size, void* d_ws, size_t ws_size,
                              hipStream_t stream) {
    const float* u       = (const float*)d_in[0];
    const float* mask    = (const float*)d_in[1];
    const float* info    = (const float*)d_in[2];
    const float* W_C     = (const float*)d_in[3];
    const float* W_B     = (const float*)d_in[4];
    const float* W_delta = (const float*)d_in[5];
    const float* W_dt    = (const float*)d_in[6];
    const float* dt_bias = (const float*)d_in[7];
    const float* A_log   = (const float*)d_in[8];
    float* out = (float*)d_out;

    float* ws = (float*)d_ws;
    size_t off = 0;
    float* dd         = ws + off; off += (size_t)B_ * L_ * D_ * 2;    // 64 MB
    float* BmT        = ws + off; off += (size_t)B_ * L_ * N_;
    float* CmT        = ws + off; off += (size_t)B_ * L_ * N_;
    float* chunkState = ws + off; off += (size_t)B_ * NCH * D_ * N_;
    float* initState  = ws + off; off += (size_t)B_ * NCH * D_ * N_;
    float* chunkSum   = ws + off; off += (size_t)B_ * NCH * D_;

    s6_proj_kernel<<<dim3(L_ / 16, B_), 256, 0, stream>>>(
        u, mask, info, W_C, W_B, W_delta, W_dt, dt_bias, dd, BmT, CmT);
    s6_scan_partial<<<dim3(D_ / 256, B_, NCH), 256, 0, stream>>>(
        dd, BmT, A_log, chunkState, chunkSum);
    s6_scan_combine<<<dim3(D_ / 256, B_), 256, 0, stream>>>(
        chunkState, chunkSum, A_log, initState);
    s6_scan_final<<<dim3(D_ / 256, B_, NCH), 256, 0, stream>>>(
        dd, BmT, CmT, A_log, initState, out);
}
